// StabilizedSMM_86998857547869
// MI455X (gfx1250) — matrix operand order repaired and checked by reading, not compiled, not run on hardware
//
#include <hip/hip_runtime.h>
#include <hip/hip_bf16.h>

#define B_ 4
#define N_ 2048
#define D_ 256
#define S_ 8
#define H_ 1024
#define EPS_ 1e-10f
#define LNE_ 1e-5f
#define SCALE_ 0.0625f   // 256^-0.5

typedef __attribute__((ext_vector_type(16))) _Float16 v16h;
typedef __attribute__((ext_vector_type(8)))  float    v8f;

static __device__ __forceinline__ float blk_reduce256(float v, float* sbuf) {
    int tid = threadIdx.x;
    sbuf[tid] = v;
    __syncthreads();
    for (int off = 128; off > 0; off >>= 1) {
        if (tid < off) sbuf[tid] += sbuf[tid + off];
        __syncthreads();
    }
    float r = sbuf[0];
    __syncthreads();
    return r;
}

// ---------------- elementwise helpers ----------------
__global__ void copy_kernel(const float* __restrict__ src, float* __restrict__ dst, int n) {
    int i = blockIdx.x * blockDim.x + threadIdx.x;
    if (i < n) dst[i] = src[i];
}
__global__ void fill_kernel(float* __restrict__ dst, float v, int n) {
    int i = blockIdx.x * blockDim.x + threadIdx.x;
    if (i < n) dst[i] = v;
}

// LayerNorm over rows of 256 (one block per row)
__global__ void ln_rows_kernel(const float* __restrict__ in, const float* __restrict__ g,
                               const float* __restrict__ b, float* __restrict__ out) {
    __shared__ float red[256];
    int row = blockIdx.x, tid = threadIdx.x;
    float v = in[(size_t)row * D_ + tid];
    float m = blk_reduce256(v, red) * (1.0f / D_);
    float d = v - m;
    float var = blk_reduce256(d * d, red) * (1.0f / D_);
    out[(size_t)row * D_ + tid] = d * rsqrtf(var + LNE_) * g[tid] + b[tid];
}

// ---------------- WMMA GEMM: C[M,N] = A[M,K] @ W[N,K]^T (f32 in/out, f16 MACs) ----
__global__ void wmma_gemm_abT_kernel(const float* __restrict__ A, const float* __restrict__ W,
                                     float* __restrict__ C, int M, int N, int K) {
    int wid = blockIdx.x * (blockDim.x >> 5) + (threadIdx.x >> 5);
    int tilesN = N >> 4;
    int totalTiles = (M >> 4) * tilesN;
    if (wid >= totalTiles) return;
    int tm = wid / tilesN, tn = wid % tilesN;
    int lane = threadIdx.x & 31, half = lane >> 4, l16 = lane & 15;
    v8f acc = {};
    int arow = tm * 16 + l16;
    int bcol = tn * 16 + l16;
    for (int kc = 0; kc < K; kc += 32) {
        v16h a, b;
        const float* ap = A + (size_t)arow * K + kc + half * 8;
#pragma unroll
        for (int i = 0; i < 8; ++i) {
            a[i]     = (_Float16)ap[i];
            a[i + 8] = (_Float16)ap[i + 16];
        }
        const float* wp = W + (size_t)bcol * K + kc + half * 8;
#pragma unroll
        for (int i = 0; i < 8; ++i) { b[i] = (_Float16)wp[i]; b[i + 8] = (_Float16)wp[i + 16]; }
        acc = __builtin_amdgcn_wmma_f32_16x16x32_f16(false, a, false, b, (short)0, acc, false, false);
    }
#pragma unroll
    for (int r = 0; r < 8; ++r) {
        int row = tm * 16 + half * 8 + r;
        C[(size_t)row * N + bcol] = acc[r];
    }
}

// ---------------- dots[b,s,n] = SCALE*(c0[b,s] - 2*k.w1 + k^2.w2) via 2 WMMAs/chunk ----
__global__ void wmma_dots_kernel(const float* __restrict__ Kmat, const float* __restrict__ w1,
                                 const float* __restrict__ w2, const float* __restrict__ c0,
                                 float* __restrict__ dots) {
    int b = blockIdx.y;
    int tm = blockIdx.x * (blockDim.x >> 5) + (threadIdx.x >> 5); // 0..127
    int lane = threadIdx.x & 31, half = lane >> 4, l16 = lane & 15;
    v8f acc1 = {}, acc2 = {};
    int nrow = tm * 16 + l16;
    const float* kbase = Kmat + ((size_t)b * N_ + nrow) * D_;
    bool svalid = (l16 < S_);
    int sidx = b * S_ + (svalid ? l16 : 0);
    const float* w1p = w1 + (size_t)sidx * D_;
    const float* w2p = w2 + (size_t)sidx * D_;
    for (int kc = 0; kc < D_; kc += 32) {
        v16h a1, a2, bb1, bb2;
        const float* ap = kbase + kc + half * 8;
#pragma unroll
        for (int i = 0; i < 8; ++i) {
            float v0 = ap[i], v1 = ap[i + 16];
            a1[i] = (_Float16)v0;          a1[i + 8] = (_Float16)v1;
            a2[i] = (_Float16)(v0 * v0);   a2[i + 8] = (_Float16)(v1 * v1);
        }
#pragma unroll
        for (int i = 0; i < 16; ++i) {
            float w1v = svalid ? w1p[kc + half * 8 + (i & 7) + ((i >> 3) << 4)] : 0.f;
            float w2v = svalid ? w2p[kc + half * 8 + (i & 7) + ((i >> 3) << 4)] : 0.f;
            bb1[i] = (_Float16)w1v;
            bb2[i] = (_Float16)w2v;
        }
        acc1 = __builtin_amdgcn_wmma_f32_16x16x32_f16(false, a1, false, bb1, (short)0, acc1, false, false);
        acc2 = __builtin_amdgcn_wmma_f32_16x16x32_f16(false, a2, false, bb2, (short)0, acc2, false, false);
    }
    if (svalid) {
        float c = c0[b * S_ + l16];
        float* dp = dots + (size_t)(b * S_ + l16) * N_ + tm * 16 + half * 8;
#pragma unroll
        for (int r = 0; r < 8; ++r)
            dp[r] = SCALE_ * (c - 2.f * acc1[r] + acc2[r]);
    }
}

// ------- out[b,s,d] = sum_n a[b,s,n]*V[b,n,d]  (a = (g+eps)*scale); optional V^2 pass ----
__global__ void wmma_slotgemm_kernel(const float* __restrict__ g, const float* __restrict__ scaleA,
                                     const float* __restrict__ V, float* __restrict__ out1,
                                     float* __restrict__ out2, int withsq) {
    int b = blockIdx.y;
    int tn = blockIdx.x * (blockDim.x >> 5) + (threadIdx.x >> 5); // 0..15
    int lane = threadIdx.x & 31, half = lane >> 4, l16 = lane & 15;
    v8f acc1 = {}, acc2 = {};
    bool svalid = (l16 < S_);
    int sidx = b * S_ + (svalid ? l16 : 0);
    const float* gp = g + (size_t)sidx * N_;
    float sc = svalid ? scaleA[b * S_ + l16] : 0.f;
    int dcol = tn * 16 + l16;
    for (int kc = 0; kc < N_; kc += 32) {
        v16h a, bv, bv2;
        int abase = kc + half * 8;
#pragma unroll
        for (int i = 0; i < 8; ++i) {
            a[i]     = (_Float16)((gp[abase + i]      + EPS_) * sc);
            a[i + 8] = (_Float16)((gp[abase + 16 + i] + EPS_) * sc);
        }
        const float* vp = V + ((size_t)b * N_ + kc + half * 8) * D_ + dcol;
#pragma unroll
        for (int i = 0; i < 16; ++i) {
            float vv = vp[(size_t)((i & 7) + ((i >> 3) << 4)) * D_];
            bv[i]  = (_Float16)vv;
            bv2[i] = (_Float16)(vv * vv);
        }
        acc1 = __builtin_amdgcn_wmma_f32_16x16x32_f16(false, a, false, bv, (short)0, acc1, false, false);
        if (withsq)
            acc2 = __builtin_amdgcn_wmma_f32_16x16x32_f16(false, a, false, bv2, (short)0, acc2, false, false);
    }
    if (half == 0) { // rows 0..7 == slots 0..7
#pragma unroll
        for (int r = 0; r < 8; ++r) {
            out1[(size_t)(b * S_ + r) * D_ + dcol] = acc1[r];
            if (withsq) out2[(size_t)(b * S_ + r) * D_ + dcol] = acc2[r];
        }
    }
}

// ------- double-norm pass: g = (de+eps)/(sum_S de+eps), de = exp(-dots/2+eps)*pi -------
__global__ void dblnorm_kernel(const float* __restrict__ dots, const float* __restrict__ piIn,
                               float* __restrict__ gOut, float* __restrict__ sumOut) {
    __shared__ float ssum[S_];
    int b = blockIdx.y;
    int n = blockIdx.x * blockDim.x + threadIdx.x;
    if (threadIdx.x < S_) ssum[threadIdx.x] = 0.f;
    __syncthreads();
    float de[S_], tot = 0.f;
#pragma unroll
    for (int s = 0; s < S_; ++s) {
        float d = dots[(size_t)(b * S_ + s) * N_ + n];
        de[s] = expf(-0.5f * d + EPS_) * piIn[b * S_ + s];
        tot += de[s];
    }
    float inv = 1.f / (tot + EPS_);
#pragma unroll
    for (int s = 0; s < S_; ++s) {
        float gg = (de[s] + EPS_) * inv;
        if (gOut) gOut[(size_t)(b * S_ + s) * N_ + n] = gg;
        atomicAdd(&ssum[s], gg);
    }
    __syncthreads();
    if (threadIdx.x < S_) atomicAdd(&sumOut[b * S_ + threadIdx.x], ssum[threadIdx.x]);
}

// pi_out[b,s] = (p+eps)/(sum_S p+eps), p = (sumg+N*eps)/(sumg+eps); scale=1/(sumg+eps)
__global__ void finpi_kernel(const float* __restrict__ sumg, float* __restrict__ piOut,
                             float* __restrict__ scaleOut, float* __restrict__ SaOut) {
    __shared__ float psh[32];
    int t = threadIdx.x;
    float sg = sumg[t];
    float p = (sg + (float)N_ * EPS_) / (sg + EPS_);
    psh[t] = p;
    __syncthreads();
    int b = t >> 3;
    float tot = 0.f;
#pragma unroll
    for (int s = 0; s < 8; ++s) tot += psh[b * 8 + s];
    piOut[t] = (p + EPS_) / (tot + EPS_);
    if (scaleOut) scaleOut[t] = 1.f / (sg + EPS_);
    if (SaOut) SaOut[t] = p;
}

// ------- slot prep: LN(slots) -> mu; q=mu@Wq^T, sig=exp(ls@Wq^T); w1,w2,c0 -------
__global__ void slot_prep_kernel(const float* __restrict__ slots, const float* __restrict__ lng,
                                 const float* __restrict__ lnb, const float* __restrict__ Wq,
                                 float* __restrict__ muOut, float* __restrict__ w1,
                                 float* __restrict__ w2, float* __restrict__ c0) {
    __shared__ float sl[2 * D_];
    __shared__ float red[256];
    int bs = blockIdx.x, tid = threadIdx.x;
    float v0 = slots[(size_t)bs * 2 * D_ + tid];
    float v1 = slots[(size_t)bs * 2 * D_ + D_ + tid];
    float m = blk_reduce256(v0 + v1, red) * (1.f / (2 * D_));
    float d0 = v0 - m, d1 = v1 - m;
    float var = blk_reduce256(d0 * d0 + d1 * d1, red) * (1.f / (2 * D_));
    float inv = rsqrtf(var + LNE_);
    sl[tid]       = d0 * inv * lng[tid] + lnb[tid];
    sl[tid + D_]  = d1 * inv * lng[tid + D_] + lnb[tid + D_];
    __syncthreads();
    float qa = 0.f, qb = 0.f;
    const float* wr = Wq + (size_t)tid * D_;
    for (int e = 0; e < D_; ++e) { qa += wr[e] * sl[e]; qb += wr[e] * sl[e + D_]; }
    float iv2 = expf(-2.f * qb);              // 1/sigma^2
    muOut[(size_t)bs * D_ + tid] = sl[tid];
    w2[(size_t)bs * D_ + tid] = iv2;
    w1[(size_t)bs * D_ + tid] = qa * iv2;
    float part = blk_reduce256(qa * qa * iv2, red);
    if (tid == 0) c0[bs] = part;
}

// ------- quadratic prep: w1 = mu*inv, (w2=inv), c0 = sum mu^2*inv -------
__global__ void prep_quad_kernel(const float* __restrict__ mu, const float* __restrict__ src,
                                 int src_is_logsig, float* __restrict__ w1,
                                 float* __restrict__ w2out, float* __restrict__ c0) {
    __shared__ float red[256];
    int bs = blockIdx.x, tid = threadIdx.x;
    float inv = src_is_logsig ? expf(-2.f * src[(size_t)bs * D_ + tid]) : src[(size_t)bs * D_ + tid];
    if (w2out) w2out[(size_t)bs * D_ + tid] = inv;
    float m = mu[(size_t)bs * D_ + tid];
    w1[(size_t)bs * D_ + tid] = m * inv;
    float part = blk_reduce256(m * m * inv, red);
    if (tid == 0) c0[bs] = part;
}

// ------- GRU + LN + mu-MLP per (b,s) -------
__global__ void gru_mlp_kernel(const float* __restrict__ upd, const float* __restrict__ mu,
                               const float* __restrict__ w_ih, const float* __restrict__ w_hh,
                               const float* __restrict__ b_ih, const float* __restrict__ b_hh,
                               const float* __restrict__ m1w, const float* __restrict__ m1b,
                               const float* __restrict__ m2w, const float* __restrict__ m2b,
                               const float* __restrict__ lng, const float* __restrict__ lnb,
                               float* __restrict__ upd_mu, float* __restrict__ slots) {
    __shared__ float ur[D_], mr[D_], umu[D_], hbuf[H_], red[256];
    int bs = blockIdx.x, tid = threadIdx.x;
    ur[tid] = upd[(size_t)bs * D_ + tid];
    mr[tid] = mu[(size_t)bs * D_ + tid];
    __syncthreads();
    float ir = b_ih[tid], iz = b_ih[D_ + tid], in_ = b_ih[2 * D_ + tid];
    float hr = b_hh[tid], hz = b_hh[D_ + tid], hn = b_hh[2 * D_ + tid];
    const float* wi0 = w_ih + (size_t)tid * D_;
    const float* wi1 = w_ih + (size_t)(D_ + tid) * D_;
    const float* wi2 = w_ih + (size_t)(2 * D_ + tid) * D_;
    const float* wh0 = w_hh + (size_t)tid * D_;
    const float* wh1 = w_hh + (size_t)(D_ + tid) * D_;
    const float* wh2 = w_hh + (size_t)(2 * D_ + tid) * D_;
    for (int e = 0; e < D_; ++e) {
        float u = ur[e], h = mr[e];
        ir += wi0[e] * u; iz += wi1[e] * u; in_ += wi2[e] * u;
        hr += wh0[e] * h; hz += wh1[e] * h; hn += wh2[e] * h;
    }
    float r = 1.f / (1.f + expf(-(ir + hr)));
    float z = 1.f / (1.f + expf(-(iz + hz)));
    float ng = tanhf(in_ + r * hn);
    float h = (1.f - z) * ng + z * mr[tid];
    float m = blk_reduce256(h, red) * (1.f / D_);
    float d = h - m;
    float var = blk_reduce256(d * d, red) * (1.f / D_);
    float u0 = d * rsqrtf(var + LNE_) * lng[tid] + lnb[tid];
    umu[tid] = u0;
    __syncthreads();
    for (int j = tid; j < H_; j += 256) {
        float a = m1b[j];
        const float* wr = m1w + (size_t)j * D_;
        for (int e = 0; e < D_; ++e) a += wr[e] * umu[e];
        hbuf[j] = fmaxf(a, 0.f);
    }
    __syncthreads();
    float o = m2b[tid];
    const float* wr2 = m2w + (size_t)tid * H_;
    for (int e = 0; e < H_; ++e) o += wr2[e] * hbuf[e];
    o += umu[tid];
    upd_mu[(size_t)bs * D_ + tid] = o;
    slots[(size_t)bs * 2 * D_ + tid] = o;
}

// ------- upd_logsig = 0.5*log(E[v^2]-2mu*E[v]+mu^2*Sa + eps) -------
__global__ void logsig_fin_kernel(const float* __restrict__ Ev, const float* __restrict__ Ev2,
                                  const float* __restrict__ upd_mu, const float* __restrict__ Sa,
                                  float* __restrict__ upd_ls, float* __restrict__ slots) {
    int bs = blockIdx.x, tid = threadIdx.x;
    float m = upd_mu[(size_t)bs * D_ + tid];
    float val = Ev2[(size_t)bs * D_ + tid] - 2.f * m * Ev[(size_t)bs * D_ + tid] + m * m * Sa[bs];
    float ls = 0.5f * logf(val + EPS_);
    upd_ls[(size_t)bs * D_ + tid] = ls;
    slots[(size_t)bs * 2 * D_ + D_ + tid] = ls;
}

// ------- last-step gamma: normalize de4 over S -------
__global__ void gam_kernel(const float* __restrict__ dots, const float* __restrict__ pi3,
                           float* __restrict__ attnOut) {
    int b = blockIdx.y;
    int n = blockIdx.x * blockDim.x + threadIdx.x;
    float de[S_], tot = 0.f;
#pragma unroll
    for (int s = 0; s < S_; ++s) {
        de[s] = expf(-0.5f * dots[(size_t)(b * S_ + s) * N_ + n] + EPS_) * pi3[b * S_ + s];
        tot += de[s];
    }
    float inv = 1.f / (tot + EPS_);
#pragma unroll
    for (int s = 0; s < S_; ++s)
        attnOut[(size_t)(b * S_ + s) * N_ + n] = (de[s] + EPS_) * inv;
}

// ------- output MLP per (b,s) -------
__global__ void final_mlp_kernel(const float* __restrict__ slots, const float* __restrict__ w1,
                                 const float* __restrict__ b1, const float* __restrict__ w2,
                                 const float* __restrict__ b2, float* __restrict__ outSlots) {
    __shared__ float inb[2 * D_];
    __shared__ float hbuf[2 * H_];
    int bs = blockIdx.x, tid = threadIdx.x;
    inb[tid] = slots[(size_t)bs * 2 * D_ + tid];
    inb[tid + D_] = slots[(size_t)bs * 2 * D_ + D_ + tid];
    __syncthreads();
    for (int j = tid; j < 2 * H_; j += 256) {
        float a = b1[j];
        const float* wr = w1 + (size_t)j * (2 * D_);
        for (int e = 0; e < 2 * D_; ++e) a += wr[e] * inb[e];
        hbuf[j] = fmaxf(a, 0.f);
    }
    __syncthreads();
    float o = b2[tid];
    const float* wr2 = w2 + (size_t)tid * (2 * H_);
    for (int e = 0; e < 2 * H_; ++e) o += wr2[e] * hbuf[e];
    outSlots[(size_t)bs * D_ + tid] = o;
}

extern "C" void kernel_launch(void* const* d_in, const int* in_sizes, int n_in,
                              void* d_out, int out_size, void* d_ws, size_t ws_size,
                              hipStream_t stream) {
    (void)in_sizes; (void)n_in; (void)out_size; (void)ws_size;
    const float* slots_in  = (const float*)d_in[0];
    const float* inputs    = (const float*)d_in[1];
    const float* Wq        = (const float*)d_in[2];
    const float* Wk        = (const float*)d_in[3];
    const float* Wv        = (const float*)d_in[4];
    const float* gru_w_ih  = (const float*)d_in[5];
    const float* gru_w_hh  = (const float*)d_in[6];
    const float* gru_b_ih  = (const float*)d_in[7];
    const float* gru_b_hh  = (const float*)d_in[8];
    const float* mlp_mu_w1 = (const float*)d_in[9];
    const float* mlp_mu_b1 = (const float*)d_in[10];
    const float* mlp_mu_w2 = (const float*)d_in[11];
    const float* mlp_mu_b2 = (const float*)d_in[12];
    const float* ln_in_g   = (const float*)d_in[13];
    const float* ln_in_b   = (const float*)d_in[14];
    const float* ln_sl_g   = (const float*)d_in[15];
    const float* ln_sl_b   = (const float*)d_in[16];
    const float* ln_mu_g   = (const float*)d_in[17];
    const float* ln_mu_b   = (const float*)d_in[18];
    const float* out_w1    = (const float*)d_in[19];
    const float* out_b1    = (const float*)d_in[20];
    const float* out_w2    = (const float*)d_in[21];
    const float* out_b2    = (const float*)d_in[22];

    float* ws = (float*)d_ws;
    float* x      = ws;                      // B*N*D
    float* kbuf   = x + (size_t)B_*N_*D_;
    float* vbuf   = kbuf + (size_t)B_*N_*D_;
    float* dots   = vbuf + (size_t)B_*N_*D_; // B*S*N
    float* g      = dots + (size_t)B_*S_*N_;
    float* g3     = g + (size_t)B_*S_*N_;
    float* mu     = g3 + (size_t)B_*S_*N_;   // B*S*D each below
    float* w1b    = mu + B_*S_*D_;
    float* w2b    = w1b + B_*S_*D_;
    float* upd    = w2b + B_*S_*D_;
    float* upd_mu = upd + B_*S_*D_;
    float* upd_ls = upd_mu + B_*S_*D_;
    float* Ev     = upd_ls + B_*S_*D_;
    float* Ev2    = Ev + B_*S_*D_;
    float* slots  = Ev2 + B_*S_*D_;          // B*S*2D
    float* c0     = slots + (size_t)B_*S_*2*D_;
    float* pi     = c0 + 32;
    float* pi1    = pi + 32;
    float* pi2    = pi1 + 32;
    float* sums   = pi2 + 32;                // sumg | sumg2 | sumg3 (3*32)
    float* scaleA = sums + 96;
    float* scale3 = scaleA + 32;
    float* Sa     = scale3 + 32;

    float* out_slots = (float*)d_out;
    float* out_attn  = out_slots + (size_t)B_*S_*D_;

    // init
    copy_kernel<<<64, 256, 0, stream>>>(slots_in, slots, B_*S_*2*D_);
    fill_kernel<<<1, 32, 0, stream>>>(pi, 1.0f / S_, 32);

    // x = LN(inputs); k = x@Wk^T; v = x@Wv^T   (WMMA)
    ln_rows_kernel<<<B_*N_, 256, 0, stream>>>(inputs, ln_in_g, ln_in_b, x);
    wmma_gemm_abT_kernel<<<1024, 256, 0, stream>>>(x, Wk, kbuf, B_*N_, D_, D_);
    wmma_gemm_abT_kernel<<<1024, 256, 0, stream>>>(x, Wv, vbuf, B_*N_, D_, D_);

    for (int it = 0; it < 4; ++it) {
        bool last = (it == 3);
        slot_prep_kernel<<<32, 256, 0, stream>>>(slots, ln_sl_g, ln_sl_b, Wq, mu, w1b, w2b, c0);
        wmma_dots_kernel<<<dim3(32, B_), 128, 0, stream>>>(kbuf, w1b, w2b, c0, dots);
        fill_kernel<<<1, 96, 0, stream>>>(sums, 0.f, 96);
        // first double-norm -> g (attn base), pi1
        dblnorm_kernel<<<dim3(8, B_), 256, 0, stream>>>(dots, pi, g, sums);
        finpi_kernel<<<1, 32, 0, stream>>>(sums, pi1, scaleA, nullptr);
        // second double-norm (sums only) -> pi2
        dblnorm_kernel<<<dim3(8, B_), 256, 0, stream>>>(dots, pi1, nullptr, sums + 32);
        finpi_kernel<<<1, 32, 0, stream>>>(sums + 32, pi2, nullptr, nullptr);
        // upd = attn @ v   (WMMA)
        wmma_slotgemm_kernel<<<dim3(4, B_), 128, 0, stream>>>(g, scaleA, vbuf, upd, nullptr, 0);
        gru_mlp_kernel<<<32, 256, 0, stream>>>(upd, mu, gru_w_ih, gru_w_hh, gru_b_ih, gru_b_hh,
                                               mlp_mu_w1, mlp_mu_b1, mlp_mu_w2, mlp_mu_b2,
                                               ln_mu_g, ln_mu_b, upd_mu, slots);
        // ndots with updated mu, same sigma
        prep_quad_kernel<<<32, 256, 0, stream>>>(upd_mu, w2b, 0, w1b, nullptr, c0);
        wmma_dots_kernel<<<dim3(32, B_), 128, 0, stream>>>(kbuf, w1b, w2b, c0, dots);
        // third double-norm -> g3, pi3 (becomes next pi), scale3, Sa
        dblnorm_kernel<<<dim3(8, B_), 256, 0, stream>>>(dots, pi2, g3, sums + 64);
        finpi_kernel<<<1, 32, 0, stream>>>(sums + 64, pi, scale3, Sa);
        // E[v], E[v^2] under a3   (WMMA)
        wmma_slotgemm_kernel<<<dim3(4, B_), 128, 0, stream>>>(g3, scale3, vbuf, Ev, Ev2, 1);
        logsig_fin_kernel<<<32, 256, 0, stream>>>(Ev, Ev2, upd_mu, Sa, upd_ls, slots);
        if (last) {
            // d2 with new sigma = exp(upd_logsig); gamma -> output attn
            prep_quad_kernel<<<32, 256, 0, stream>>>(upd_mu, upd_ls, 1, w1b, w2b, c0);
            wmma_dots_kernel<<<dim3(32, B_), 128, 0, stream>>>(kbuf, w1b, w2b, c0, dots);
            gam_kernel<<<dim3(8, B_), 256, 0, stream>>>(dots, pi, out_attn);
        }
    }
    final_mlp_kernel<<<32, 256, 0, stream>>>(slots, out_w1, out_b1, out_w2, out_b2, out_slots);
}
